// Net_45148696216268
// MI455X (gfx1250) — matrix operand order repaired and checked by reading, not compiled, not run on hardware
//
#include <hip/hip_runtime.h>
#include <hip/hip_bf16.h>
#include <stdint.h>

typedef __bf16       v16bf __attribute__((ext_vector_type(16)));
typedef float        v8f   __attribute__((ext_vector_type(8)));
typedef unsigned int v8u   __attribute__((ext_vector_type(8)));

#define EMB        128
#define K_CODES    256
#define LAYERS     2
#define BLOCK      512
#define WAVES      (BLOCK / 32)
#define TOK_PER_WG (WAVES * 16)   // 256 tokens per workgroup

__device__ __forceinline__ unsigned short bfbits(float f) {
  unsigned u = __builtin_bit_cast(unsigned, f);
  u += 0x7FFFu + ((u >> 16) & 1u);            // round-to-nearest-even
  return (unsigned short)(u >> 16);
}
__device__ __forceinline__ __bf16 tobf(float f) {
  unsigned short s = bfbits(f);
  return __builtin_bit_cast(__bf16, s);
}
__device__ __forceinline__ float frombf(__bf16 b) {
  unsigned short s = __builtin_bit_cast(unsigned short, b);
  unsigned u = ((unsigned)s) << 16;
  return __builtin_bit_cast(float, u);
}
__device__ __forceinline__ unsigned pack2bf(float a, float b) {
  return (unsigned)bfbits(a) | ((unsigned)bfbits(b) << 16);
}
// Monotonic total-order mapping: f32 -> u32 s.t. a < b  <=>  map(a) < map(b)
__device__ __forceinline__ unsigned monokey(float f) {
  unsigned u = __builtin_bit_cast(unsigned, f);
  return (u & 0x80000000u) ? ~u : (u | 0x80000000u);
}

__global__ __launch_bounds__(BLOCK) void rvq_fused_kernel(
    const float* __restrict__ x, const float* __restrict__ cb,
    float* __restrict__ outq, int* __restrict__ idxOut, int tokens)
{
  // bf16 codebook pre-swizzled into WMMA-B fragment order:
  // [layer][ntile(16)][kstep(4)][lane(32)][8 dwords] = 128 KB
  __shared__ unsigned cbb[LAYERS * 16 * 4 * 32 * 8];
  __shared__ float    c2s[LAYERS * K_CODES];      // ||c||^2, 2 KB

  const int tid  = threadIdx.x;
  const int lane = tid & 31;
  const int wave = tid >> 5;

  // ---- preload: c2 = ||c||^2 (one code per thread; BLOCK == LAYERS*K_CODES) ----
  {
    const float4* row = (const float4*)(cb + (size_t)tid * EMB);
    float s = 0.f;
    #pragma unroll
    for (int i = 0; i < EMB / 4; ++i) {
      float4 v = row[i];
      s += v.x * v.x + v.y * v.y + v.z * v.z + v.w * v.w;
    }
    c2s[tid] = s;
  }

  // ---- preload: bf16 B fragments (4096 16-dim chunks, 8 per thread) ----
  #pragma unroll
  for (int j = 0; j < 8; ++j) {
    int chunk = j * BLOCK + tid;
    int layer = chunk >> 11;            // 2048 chunks per layer
    int code  = (chunk >> 3) & 255;
    int kc    = chunk & 7;              // 8 x 16-dim chunks per code
    int ks    = kc >> 1;
    int half  = kc & 1;
    int nt    = code >> 4;
    int laneSlot = (code & 15) + (half << 4);   // B layout: lanes 0-15 K=0..7,16..23, 16-31 K=8..15,24..31
    const float* src = cb + (((size_t)(layer << 8) + code) << 7) + (ks << 5) + (half << 3);
    float4 f0 = *(const float4*)(src);
    float4 f1 = *(const float4*)(src + 4);
    float4 f2 = *(const float4*)(src + 16);
    float4 f3 = *(const float4*)(src + 20);
    uint4 o0 = { pack2bf(f0.x, f0.y), pack2bf(f0.z, f0.w),
                 pack2bf(f1.x, f1.y), pack2bf(f1.z, f1.w) };
    uint4 o1 = { pack2bf(f2.x, f2.y), pack2bf(f2.z, f2.w),
                 pack2bf(f3.x, f3.y), pack2bf(f3.z, f3.w) };
    unsigned* d = &cbb[(size_t)((((layer * 16 + nt) * 4 + ks) * 32 + laneSlot) * 8)];
    ((uint4*)d)[0] = o0;
    ((uint4*)d)[1] = o1;
  }
  __syncthreads();

  // ---- per-wave tile of 16 tokens; residual in registers in A-fragment order ----
  // lane L holds row (L&15); dims per k-step s: {32s+b..32s+b+7, 32s+b+16..32s+b+23},
  // b = 0 for lanes 0-15, b = 8 for lanes 16-31 (matches 16-bit A 16x32 VGPR layout).
  const int rowBase = blockIdx.x * TOK_PER_WG + wave * 16;
  const int myRow   = rowBase + (lane & 15);
  const int base8   = (lane < 16) ? 0 : 8;
  const float* xrow = x + (size_t)myRow * EMB;

  float r[64];
  #pragma unroll
  for (int s = 0; s < 4; ++s) {
    float4 a0 = *(const float4*)(xrow + 32 * s + base8);
    float4 a1 = *(const float4*)(xrow + 32 * s + base8 + 4);
    float4 a2 = *(const float4*)(xrow + 32 * s + base8 + 16);
    float4 a3 = *(const float4*)(xrow + 32 * s + base8 + 20);
    r[s*16+ 0]=a0.x; r[s*16+ 1]=a0.y; r[s*16+ 2]=a0.z; r[s*16+ 3]=a0.w;
    r[s*16+ 4]=a1.x; r[s*16+ 5]=a1.y; r[s*16+ 6]=a1.z; r[s*16+ 7]=a1.w;
    r[s*16+ 8]=a2.x; r[s*16+ 9]=a2.y; r[s*16+10]=a2.z; r[s*16+11]=a2.w;
    r[s*16+12]=a3.x; r[s*16+13]=a3.y; r[s*16+14]=a3.z; r[s*16+15]=a3.w;
  }

  const int nsub = lane & 15;

  for (int l = 0; l < LAYERS; ++l) {
    // split residual into bf16 hi + lo A-fragments (near-f32 dot accuracy)
    v16bf ahi[4], alo[4];
    #pragma unroll
    for (int s = 0; s < 4; ++s) {
      #pragma unroll
      for (int i = 0; i < 16; ++i) {
        float f = r[s * 16 + i];
        __bf16 h = tobf(f);
        ahi[s][i] = h;
        alo[s][i] = tobf(f - frombf(h));
      }
    }

    float minv[8];
    int   mini[8];
    #pragma unroll
    for (int j = 0; j < 8; ++j) { minv[j] = __builtin_inff(); mini[j] = 0; }

    const unsigned* cbL = cbb + (size_t)l * (16 * 4 * 32 * 8);

    #pragma unroll 1
    for (int nt = 0; nt < 16; ++nt) {
      v8f acc = {0.f, 0.f, 0.f, 0.f, 0.f, 0.f, 0.f, 0.f};
      #pragma unroll
      for (int ks = 0; ks < 4; ++ks) {
        v8u bu = *(const v8u*)(cbL + (size_t)((nt * 4 + ks) * 32 + lane) * 8);
        v16bf b = __builtin_bit_cast(v16bf, bu);
        acc = __builtin_amdgcn_wmma_f32_16x16x32_bf16(
                  false, alo[ks], false, b, (short)0, acc, false, false);
        acc = __builtin_amdgcn_wmma_f32_16x16x32_bf16(
                  false, ahi[ks], false, b, (short)0, acc, false, false);
      }
      // score = ||c||^2 - 2 r.c  (||r||^2 constant per row -> irrelevant to argmin)
      float c2v = c2s[l * K_CODES + nt * 16 + nsub];
      int   idx = nt * 16 + nsub;
      #pragma unroll
      for (int j = 0; j < 8; ++j) {
        float sc = c2v - 2.0f * acc[j];
        if (sc < minv[j]) { minv[j] = sc; mini[j] = idx; }  // strict < keeps first min
      }
    }

    // Branchless argmin reduction across the 16 lanes of each half-wave:
    // pack (score,idx) into one monotonic u64 key; min over packed keys.
    // Ties resolve to the lowest index (matches jnp.argmin first-min).
    #pragma unroll
    for (int j = 0; j < 8; ++j) {
      unsigned long long key =
          ((unsigned long long)monokey(minv[j]) << 32) | (unsigned)mini[j];
      #pragma unroll
      for (int off = 1; off < 16; off <<= 1) {
        unsigned long long ok = __shfl_xor(key, off, 32);
        key = (ok < key) ? ok : key;
      }
      mini[j] = (int)(unsigned)(key & 0xFFFFFFFFull);
    }

    // route the winning code of row (lane&15) to this lane.
    // slot j holds row j (lanes 0-15) / row 8+j (lanes 16-31).
    int sel = mini[0];
    #pragma unroll
    for (int j = 1; j < 8; ++j) sel = ((lane & 7) == j) ? mini[j] : sel;
    int other    = __shfl_xor(sel, 16, 32);
    int needHigh = (lane >> 3) & 1;    // rows 8-15 needed?
    int haveHigh = (lane >> 4) & 1;    // this half holds rows 8-15?
    int code     = (needHigh == haveHigh) ? sel : other;

    if (lane < 16) idxOut[(size_t)l * tokens + myRow] = code;

    // residual update with exact f32 codebook row (tiny, L2-resident gather)
    const float* crow = cb + (((size_t)(l << 8) + code) << 7);
    #pragma unroll
    for (int s = 0; s < 4; ++s) {
      float4 c0 = *(const float4*)(crow + 32 * s + base8);
      float4 c1 = *(const float4*)(crow + 32 * s + base8 + 4);
      float4 c2q = *(const float4*)(crow + 32 * s + base8 + 16);
      float4 c3 = *(const float4*)(crow + 32 * s + base8 + 20);
      r[s*16+ 0]-=c0.x; r[s*16+ 1]-=c0.y; r[s*16+ 2]-=c0.z; r[s*16+ 3]-=c0.w;
      r[s*16+ 4]-=c1.x; r[s*16+ 5]-=c1.y; r[s*16+ 6]-=c1.z; r[s*16+ 7]-=c1.w;
      r[s*16+ 8]-=c2q.x; r[s*16+ 9]-=c2q.y; r[s*16+10]-=c2q.z; r[s*16+11]-=c2q.w;
      r[s*16+12]-=c3.x; r[s*16+13]-=c3.y; r[s*16+14]-=c3.z; r[s*16+15]-=c3.w;
    }
  }

  // out = x - final_residual  (== q1 + q2; x tile is hot in L2)
  float* orow = outq + (size_t)myRow * EMB;
  #pragma unroll
  for (int s = 0; s < 4; ++s) {
    float4 a0 = *(const float4*)(xrow + 32 * s + base8);
    float4 a1 = *(const float4*)(xrow + 32 * s + base8 + 4);
    float4 a2 = *(const float4*)(xrow + 32 * s + base8 + 16);
    float4 a3 = *(const float4*)(xrow + 32 * s + base8 + 20);
    float4 q0 = { a0.x - r[s*16+ 0], a0.y - r[s*16+ 1], a0.z - r[s*16+ 2], a0.w - r[s*16+ 3] };
    float4 q1 = { a1.x - r[s*16+ 4], a1.y - r[s*16+ 5], a1.z - r[s*16+ 6], a1.w - r[s*16+ 7] };
    float4 q2 = { a2.x - r[s*16+ 8], a2.y - r[s*16+ 9], a2.z - r[s*16+10], a2.w - r[s*16+11] };
    float4 q3 = { a3.x - r[s*16+12], a3.y - r[s*16+13], a3.z - r[s*16+14], a3.w - r[s*16+15] };
    *(float4*)(orow + 32 * s + base8)      = q0;
    *(float4*)(orow + 32 * s + base8 + 4)  = q1;
    *(float4*)(orow + 32 * s + base8 + 16) = q2;
    *(float4*)(orow + 32 * s + base8 + 20) = q3;
  }
}

extern "C" void kernel_launch(void* const* d_in, const int* in_sizes, int n_in,
                              void* d_out, int out_size, void* d_ws, size_t ws_size,
                              hipStream_t stream) {
  (void)n_in; (void)out_size; (void)d_ws; (void)ws_size;
  const float* x  = (const float*)d_in[0];   // [8, 65536, 128] f32
  const float* cb = (const float*)d_in[1];   // [2, 256, 128]   f32

  const int tokens = in_sizes[0] / EMB;      // 524288 (divisible by TOK_PER_WG)
  float* outq  = (float*)d_out;              // quantized_st, tokens*128 floats
  int*   idxOut = (int*)(outq + (size_t)tokens * EMB);  // indices [2, tokens] int32

  const int grid = tokens / TOK_PER_WG;
  rvq_fused_kernel<<<grid, BLOCK, 0, stream>>>(x, cb, outq, idxOut, tokens);
}
